// UDLSLinear_86569360818824
// MI455X (gfx1250) — matrix operand order repaired and checked by reading, not compiled, not run on hardware
//
#include <hip/hip_runtime.h>

// Problem constants (from the reference)
#define IN_F   4096
#define OUT_F  4096
#define GSZ    8

// GEMM tiling: 128x256x32 block tile, 256 threads = 8 waves (wave32),
// waves arranged 2 (M) x 4 (N); each wave computes a 64x64 tile = 4x4
// 16x16 WMMA tiles (128 accumulator VGPRs). Double-buffered LDS.
#define BM 128
#define BN 256
#define BK 32

typedef __attribute__((ext_vector_type(16))) _Float16 v16h;
typedef __attribute__((ext_vector_type(8)))  _Float16 v8h;
typedef __attribute__((ext_vector_type(8)))  float    v8f;
typedef __attribute__((ext_vector_type(4)))  float    v4f;

// ---------------------------------------------------------------------------
// Phase 1: decode W[o,i] = codebook[C[g]][j]*S[g] + R[o,i]  -> f16 row-major
// One thread per group (8 elements). 16B vector load/stores throughout.
// ---------------------------------------------------------------------------
__global__ __launch_bounds__(256) void udls_decode_f16(
    const float* __restrict__ codebook,
    const int*   __restrict__ C,
    const float* __restrict__ S,
    const float* __restrict__ R,
    _Float16*    __restrict__ W)
{
    unsigned g = blockIdx.x * 256u + threadIdx.x;   // < 2^21 groups
    int   idx = C[g];
    float s   = S[g];
    const v4f* cb = (const v4f*)(codebook + (unsigned long long)idx * GSZ);
    const v4f* rp = (const v4f*)(R + (unsigned long long)g * GSZ);
    v4f c0 = cb[0], c1 = cb[1];
    v4f r0 = rp[0], r1 = rp[1];
    v8h w;
    w[0] = (_Float16)(c0.x * s + r0.x);
    w[1] = (_Float16)(c0.y * s + r0.y);
    w[2] = (_Float16)(c0.z * s + r0.z);
    w[3] = (_Float16)(c0.w * s + r0.w);
    w[4] = (_Float16)(c1.x * s + r1.x);
    w[5] = (_Float16)(c1.y * s + r1.y);
    w[6] = (_Float16)(c1.z * s + r1.z);
    w[7] = (_Float16)(c1.w * s + r1.w);
    *(v8h*)(W + (unsigned long long)g * GSZ) = w;
}

// ---------------------------------------------------------------------------
// Phase 2: out[m,n] = sum_k x[m,k] * W[n,k] + bias[n]
// A (x): global f32 -> VALU cvt -> ds_store_b128 into LDS (f16).
// B (W): global_load_async_to_lds_b128 (ASYNCcnt path), f16 direct to LDS.
// Compute: v_wmma_f32_16x16x32_f16, 16 per wave per K-step, double-buffered.
// ---------------------------------------------------------------------------
__global__ __launch_bounds__(256) void udls_gemm_wmma(
    const float*    __restrict__ X,     // [M, IN_F] f32
    const _Float16* __restrict__ W,     // [OUT_F, IN_F] f16
    const float*    __restrict__ bias,  // [OUT_F]
    float*          __restrict__ out)   // [M, OUT_F]
{
    __shared__ __align__(16) _Float16 sA[2][BM * BK];  // [m][k], 64B rows (16KB)
    __shared__ __align__(16) _Float16 sB[2][BN * BK];  // [n][k], 64B rows (32KB)

    const int tid  = threadIdx.x;
    const int lane = tid & 31;
    const int wv   = tid >> 5;       // 0..7
    const int wm   = wv >> 2;        // 0..1  (64 rows each)
    const int wn   = wv & 3;         // 0..3  (64 cols each)

    const int bm = blockIdx.y * BM;
    const int bn = blockIdx.x * BN;

    // A staging: thread t covers half a 64B LDS row (32 bytes).
    const int sRow  = tid >> 1;      // 0..127
    const int sHalf = tid & 1;       // 0 or 1
    const float* gA = X + (unsigned long long)(bm + sRow) * IN_F + sHalf * 16;
    _Float16* dA0 = &sA[0][sRow * BK + sHalf * 16];
    _Float16* dA1 = &sA[1][sRow * BK + sHalf * 16];

    // B staging: thread t owns one full 64B row (4 async b128 per tile).
    // Low 32 bits of a generic __shared__ pointer == LDS byte address.
    const unsigned ldsB0 = (unsigned)(unsigned long long)&sB[0][tid * BK];
    const unsigned ldsB1 = (unsigned)(unsigned long long)&sB[1][tid * BK];
    unsigned gOffB = (unsigned)((unsigned)(bn + tid) * IN_F *
                                (unsigned)sizeof(_Float16));

    // Fragment gather geometry (ISA 16-bit A 16x32 / B 32x16 layouts).
    const int rA   = lane & 15;      // row (A: M, B: N) within 16
    const int kgrp = lane >> 4;      // lane-half selects K interleave

    union Frag { v16h v; v8h h[2]; };

    // ---- Prologue: stage tile 0 into buffer 0 ----
    asm volatile(
        "global_load_async_to_lds_b128 %0, %1, %2 offset:0\n\t"
        "global_load_async_to_lds_b128 %0, %1, %2 offset:16\n\t"
        "global_load_async_to_lds_b128 %0, %1, %2 offset:32\n\t"
        "global_load_async_to_lds_b128 %0, %1, %2 offset:48"
        :: "v"(ldsB0), "v"(gOffB), "s"(W) : "memory");
    {
        v4f x0 = *(const v4f*)(gA + 0);
        v4f x1 = *(const v4f*)(gA + 4);
        v4f x2 = *(const v4f*)(gA + 8);
        v4f x3 = *(const v4f*)(gA + 12);
        v8h h0, h1;
        h0[0]=(_Float16)x0.x; h0[1]=(_Float16)x0.y; h0[2]=(_Float16)x0.z; h0[3]=(_Float16)x0.w;
        h0[4]=(_Float16)x1.x; h0[5]=(_Float16)x1.y; h0[6]=(_Float16)x1.z; h0[7]=(_Float16)x1.w;
        h1[0]=(_Float16)x2.x; h1[1]=(_Float16)x2.y; h1[2]=(_Float16)x2.z; h1[3]=(_Float16)x2.w;
        h1[4]=(_Float16)x3.x; h1[5]=(_Float16)x3.y; h1[6]=(_Float16)x3.z; h1[7]=(_Float16)x3.w;
        ((v8h*)dA0)[0] = h0;
        ((v8h*)dA0)[1] = h1;
    }
    gA    += BK;
    gOffB += BK * (unsigned)sizeof(_Float16);
    asm volatile("s_wait_asynccnt 0" ::: "memory");
    __syncthreads();

    v8f acc[4][4] = {};              // 128 VGPRs of f32 accumulators

    const int NSTEP = IN_F / BK;     // 128
    for (int kt = 0; kt < NSTEP; ++kt) {
        const int  cur = kt & 1;
        const bool pf  = (kt + 1) < NSTEP;

        // --- Issue prefetch of tile kt+1 into the other buffer ---
        v4f x0 = {}, x1 = {}, x2 = {}, x3 = {};
        if (pf) {
            unsigned ldsBn = cur ? ldsB0 : ldsB1;
            asm volatile(
                "global_load_async_to_lds_b128 %0, %1, %2 offset:0\n\t"
                "global_load_async_to_lds_b128 %0, %1, %2 offset:16\n\t"
                "global_load_async_to_lds_b128 %0, %1, %2 offset:32\n\t"
                "global_load_async_to_lds_b128 %0, %1, %2 offset:48"
                :: "v"(ldsBn), "v"(gOffB), "s"(W) : "memory");
            x0 = *(const v4f*)(gA + 0);
            x1 = *(const v4f*)(gA + 4);
            x2 = *(const v4f*)(gA + 8);
            x3 = *(const v4f*)(gA + 12);
            gA    += BK;
            gOffB += BK * (unsigned)sizeof(_Float16);
        }

        // --- Gather A fragments once; stream B fragments per ni ---
        // A lane L: row=L%16; halves K = kgrp*8+{0..7} then 16+kgrp*8+{0..7}
        // B lane L: col=L%16; halves K = kgrp*8+{0..7} then 16+kgrp*8+{0..7}
        const _Float16* aBuf = cur ? &sA[1][0] : &sA[0][0];
        const _Float16* bBuf = cur ? &sB[1][0] : &sB[0][0];
        Frag a[4];
        #pragma unroll
        for (int mi = 0; mi < 4; ++mi) {
            const _Float16* p = aBuf + (wm * 64 + mi * 16 + rA) * BK + kgrp * 8;
            a[mi].h[0] = *(const v8h*)(p);
            a[mi].h[1] = *(const v8h*)(p + 16);
        }
        #pragma unroll
        for (int ni = 0; ni < 4; ++ni) {
            const _Float16* p = bBuf + (wn * 64 + ni * 16 + rA) * BK + kgrp * 8;
            Frag b;
            b.h[0] = *(const v8h*)(p);
            b.h[1] = *(const v8h*)(p + 16);
            #pragma unroll
            for (int mi = 0; mi < 4; ++mi)
                acc[mi][ni] = __builtin_amdgcn_wmma_f32_16x16x32_f16(
                    false, a[mi].v, false, b.v,
                    (short)0, acc[mi][ni], false, false);
        }

        // --- Finish staging A for tile kt+1 into the other buffer ---
        if (pf) {
            v8h h0, h1;
            h0[0]=(_Float16)x0.x; h0[1]=(_Float16)x0.y; h0[2]=(_Float16)x0.z; h0[3]=(_Float16)x0.w;
            h0[4]=(_Float16)x1.x; h0[5]=(_Float16)x1.y; h0[6]=(_Float16)x1.z; h0[7]=(_Float16)x1.w;
            h1[0]=(_Float16)x2.x; h1[1]=(_Float16)x2.y; h1[2]=(_Float16)x2.z; h1[3]=(_Float16)x2.w;
            h1[4]=(_Float16)x3.x; h1[5]=(_Float16)x3.y; h1[6]=(_Float16)x3.z; h1[7]=(_Float16)x3.w;
            _Float16* dAn = cur ? dA0 : dA1;
            ((v8h*)dAn)[0] = h0;
            ((v8h*)dAn)[1] = h1;
        }

        // Async LDS writes are invisible to the compiler: wait explicitly.
        // Compiler inserts s_wait_dscnt before the barrier for the ds_stores.
        asm volatile("s_wait_asynccnt 0" ::: "memory");
        __syncthreads();
    }

    // --- Epilogue: C/D layout (VGPR r -> M=r or r+8 by lane half; N=lane%16)
    const int nLane = lane & 15;
    const int mOff  = (lane >> 4) * 8;
    #pragma unroll
    for (int ni = 0; ni < 4; ++ni) {
        int n = bn + wn * 64 + ni * 16 + nLane;
        float bv = bias[n];
        #pragma unroll
        for (int mi = 0; mi < 4; ++mi) {
            int mBase = bm + wm * 64 + mi * 16 + mOff;
            #pragma unroll
            for (int r = 0; r < 8; ++r)
                out[(unsigned long long)(mBase + r) * OUT_F + n] = acc[mi][ni][r] + bv;
        }
    }
}

// ---------------------------------------------------------------------------
extern "C" void kernel_launch(void* const* d_in, const int* in_sizes, int n_in,
                              void* d_out, int out_size, void* d_ws, size_t ws_size,
                              hipStream_t stream) {
    (void)n_in; (void)out_size; (void)ws_size;
    const float* x        = (const float*)d_in[0];   // [8,2048,4096]
    const float* codebook = (const float*)d_in[1];   // [256,8]
    const int*   C        = (const int*)  d_in[2];   // [2M]
    const float* S        = (const float*)d_in[3];   // [2M]
    const float* R        = (const float*)d_in[4];   // [4096,4096]
    const float* bias     = (const float*)d_in[5];   // [4096]

    _Float16* Wf16 = (_Float16*)d_ws;                // 32 MB scratch

    const int num_groups = (OUT_F * IN_F) / GSZ;     // 2,097,152
    udls_decode_f16<<<num_groups / 256, 256, 0, stream>>>(codebook, C, S, R, Wf16);

    const int M = in_sizes[0] / IN_F;                // 16384
    dim3 grid(OUT_F / BN, M / BM);                   // (16, 128)
    udls_gemm_wmma<<<grid, 256, 0, stream>>>(x, Wf16, bias, (float*)d_out);
}
